// PairEmbed_5729486373034
// MI455X (gfx1250) — matrix operand order repaired and checked by reading, not compiled, not run on hardware
//
#include <hip/hip_runtime.h>
#include <hip/hip_bf16.h>
#include <math.h>

typedef __attribute__((ext_vector_type(16))) _Float16 v16h;
typedef __attribute__((ext_vector_type(8)))  float    v8f;

#define Sdim   128
#define Bdim   64
#define NPAIR  8128                 // S*(S-1)/2
#define NCOLS  (Bdim * NPAIR)       // 520192 (multiple of 16 and 256)
#define PI_F   3.14159265358979323846f
#define TWOPI_F 6.28318530717958647692f

__device__ __forceinline__ float gelu_exact(float x) {
    return 0.5f * x * (1.0f + erff(x * 0.70710678118654752f));
}

__device__ __forceinline__ void pair_ij(int p, int& i, int& j) {
    int ii = (int)((1.0f + sqrtf(1.0f + 8.0f * (float)p)) * 0.5f);
    while ((ii * (ii - 1)) / 2 > p) --ii;
    while (((ii + 1) * ii) / 2 <= p) ++ii;
    i = ii;
    j = p - (ii * (ii - 1)) / 2;
}

// ---------------------------------------------------------------------------
// Zero the stats scratch region
// ---------------------------------------------------------------------------
__global__ void zero_ws_kernel(float* w, int n) {
    int idx = blockIdx.x * blockDim.x + threadIdx.x;
    if (idx < n) w[idx] = 0.0f;
}

// ---------------------------------------------------------------------------
// Pairwise features + fused BN0 statistics
// feats layout: column-major-tiled [col][4]  (col = b*NPAIR + p)
// ---------------------------------------------------------------------------
__global__ __launch_bounds__(256)
void feats_kernel(const float* __restrict__ x, float* __restrict__ feats,
                  float* __restrict__ sums, float* __restrict__ sqsums) {
    const int idx = blockIdx.x * 256 + threadIdx.x;      // 0 .. NCOLS-1 exactly
    const int b = idx / NPAIR;
    const int p = idx - b * NPAIR;
    int i, j; pair_ij(p, i, j);

    const float* xb = x + (size_t)b * 4 * Sdim;
    const float pxi = xb[0 * Sdim + i], pyi = xb[1 * Sdim + i];
    const float pzi = xb[2 * Sdim + i], ei  = xb[3 * Sdim + i];
    const float pxj = xb[0 * Sdim + j], pyj = xb[1 * Sdim + j];
    const float pzj = xb[2 * Sdim + j], ej  = xb[3 * Sdim + j];

    const float pti = sqrtf(pxi * pxi + pyi * pyi);
    const float ptj = sqrtf(pxj * pxj + pyj * pyj);
    const float rapi = 0.5f * log1pf(2.0f * pzi / fmaxf(ei - pzi, 1e-20f));
    const float rapj = 0.5f * log1pf(2.0f * pzj / fmaxf(ej - pzj, 1e-20f));
    const float phii = atan2f(pyi, pxi);
    const float phij = atan2f(pyj, pxj);

    float d = phii - phij + PI_F;
    float m = fmodf(d, TWOPI_F);
    if (m < 0.0f) m += TWOPI_F;
    const float dphi = m - PI_F;
    const float drap = rapi - rapj;
    const float delta = sqrtf(drap * drap + dphi * dphi);

    const float lndelta = logf(fmaxf(delta, 1e-8f));
    const float ptmin = fminf(pti, ptj);
    const float lnkt = logf(fmaxf(ptmin * delta, 1e-8f));
    const float lnz  = logf(fmaxf(ptmin / fmaxf(pti + ptj, 1e-8f), 1e-8f));

    const float sx = pxi + pxj, sy = pyi + pyj, sz = pzi + pzj, se = ei + ej;
    const float m2 = fmaxf(se * se - (sx * sx + sy * sy + sz * sz), 1e-8f);
    const float lnm2 = logf(m2);

    float f[4] = {lnkt, lnz, lndelta, lnm2};
    float4 f4; f4.x = f[0]; f4.y = f[1]; f4.z = f[2]; f4.w = f[3];
    ((float4*)feats)[idx] = f4;                           // one b128 store

    // wave32 butterfly reduce per channel, lane0 does the global atomic
    const int lane = threadIdx.x & 31;
#pragma unroll
    for (int c = 0; c < 4; ++c) {
        float s = f[c], q = f[c] * f[c];
#pragma unroll
        for (int mask = 1; mask < 32; mask <<= 1) {
            s += __shfl_xor(s, mask);
            q += __shfl_xor(q, mask);
        }
        if (lane == 0) {
            atomicAdd(&sums[c], s);
            atomicAdd(&sqsums[c], q);
        }
    }
}

// ---------------------------------------------------------------------------
// BN finalize: (sum, sumsq) -> (scale, shift)
// ---------------------------------------------------------------------------
__global__ void finalize_bn_kernel(const float* __restrict__ sums,
                                   const float* __restrict__ sqsums,
                                   const float* __restrict__ gamma,
                                   const float* __restrict__ beta,
                                   float* __restrict__ scale,
                                   float* __restrict__ shift,
                                   int C, float invN) {
    int c = threadIdx.x;
    if (c < C) {
        float mean = sums[c] * invN;
        float var  = sqsums[c] * invN - mean * mean;
        float sc   = gamma[c] * rsqrtf(var + 1e-5f);
        scale[c] = sc;
        shift[c] = beta[c] - mean * sc;
    }
}

// ---------------------------------------------------------------------------
// Fused: [input BN-affine (+GELU)] -> GEMM (W: MOUT x KDIM) via WMMA f16
//        -> +bias -> store pre-BN output + fused per-channel sum/sumsq stats
// Activation layout: [col][C]  (C contiguous per column -> b128 load/store)
// ---------------------------------------------------------------------------
template <int KDIM, bool GELU_IN, int MOUT>
__global__ __launch_bounds__(256)
void gemm_bn_stats_kernel(const float* __restrict__ W,      // [MOUT][KDIM]
                          const float* __restrict__ bias,   // [MOUT]
                          const float* __restrict__ scale,  // [KDIM]
                          const float* __restrict__ shift,  // [KDIM]
                          const float* __restrict__ actin,  // [N][KDIM]
                          float* __restrict__ actout,       // [N][MOUT]
                          float* __restrict__ sums,         // [MOUT]
                          float* __restrict__ sqsums) {     // [MOUT]
    constexpr int KCH = (KDIM + 31) / 32;   // K chunks of 32
    constexpr int MT  = (MOUT + 15) / 16;   // 16-row m tiles

    const int lane  = threadIdx.x & 31;
    const int wid   = threadIdx.x >> 5;
    const int gwave = blockIdx.x * 8 + wid;
    const int nwav  = gridDim.x * 8;
    const int l15   = lane & 15;
    const bool hi   = lane >= 16;

    // --- A fragments (weights, f32 -> f16), contiguous k order: k = 16*hi + h ---
    v16h a[MT][KCH];
#pragma unroll
    for (int mt = 0; mt < MT; ++mt) {
#pragma unroll
        for (int kc = 0; kc < KCH; ++kc) {
            const int mrow = mt * 16 + l15;
#pragma unroll
            for (int h = 0; h < 16; ++h) {
                const int k = kc * 32 + h + (hi ? 16 : 0);
                float v = 0.0f;
                if (mrow < MOUT && k < KDIM) v = W[mrow * KDIM + k];
                a[mt][kc][h] = (_Float16)v;
            }
        }
    }

    float s_sum[MT][8], s_sq[MT][8];
#pragma unroll
    for (int mt = 0; mt < MT; ++mt)
#pragma unroll
        for (int r = 0; r < 8; ++r) { s_sum[mt][r] = 0.0f; s_sq[mt][r] = 0.0f; }

    const int ntiles = NCOLS / 16;
    for (int t = gwave; t < ntiles; t += nwav) {
        const float* colbase = actin + ((size_t)t * 16 + l15) * KDIM;

        // --- B fragments: contiguous k-run per lane -> b128 loads ---
        v16h bfr[KCH];
#pragma unroll
        for (int kc = 0; kc < KCH; ++kc) {
            const int kb = kc * 32 + (hi ? 16 : 0);
            float tmp[16];
            if (kb + 16 <= KDIM) {
                const float4* p4 = (const float4*)(colbase + kb);
#pragma unroll
                for (int q = 0; q < 4; ++q) {
                    float4 v4 = p4[q];
                    tmp[q * 4 + 0] = v4.x; tmp[q * 4 + 1] = v4.y;
                    tmp[q * 4 + 2] = v4.z; tmp[q * 4 + 3] = v4.w;
                }
            } else {
#pragma unroll
                for (int h = 0; h < 16; ++h) {
                    const int k = kb + h;
                    tmp[h] = (k < KDIM) ? colbase[k] : 0.0f;
                }
            }
#pragma unroll
            for (int h = 0; h < 16; ++h) {
                const int k = kb + h;
                float v = 0.0f;
                if (k < KDIM) {
                    v = tmp[h] * scale[k] + shift[k];
                    if (GELU_IN) v = gelu_exact(v);
                }
                bfr[kc][h] = (_Float16)v;
            }
        }

        // --- WMMA accumulate over K, then bias + b128 stores + stats ---
#pragma unroll
        for (int mt = 0; mt < MT; ++mt) {
            v8f c = {0.f, 0.f, 0.f, 0.f, 0.f, 0.f, 0.f, 0.f};
#pragma unroll
            for (int kc = 0; kc < KCH; ++kc) {
                c = __builtin_amdgcn_wmma_f32_16x16x32_f16(
                        false, a[mt][kc], false, bfr[kc],
                        (short)0, c, false, false);
            }
            const int mbase = mt * 16 + (hi ? 8 : 0);   // 8 consecutive channels
            float vals[8];
#pragma unroll
            for (int r = 0; r < 8; ++r) {
                const int mrow = mbase + r;
                float v = 0.0f;
                if (mrow < MOUT) {
                    v = c[r] + bias[mrow];
                    s_sum[mt][r] += v;
                    s_sq[mt][r]  += v * v;
                }
                vals[r] = v;
            }
            if (mbase + 8 <= MOUT) {
                float* op = actout + ((size_t)t * 16 + l15) * MOUT + mbase;
                float4 lo4, hi4;
                lo4.x = vals[0]; lo4.y = vals[1]; lo4.z = vals[2]; lo4.w = vals[3];
                hi4.x = vals[4]; hi4.y = vals[5]; hi4.z = vals[6]; hi4.w = vals[7];
                ((float4*)op)[0] = lo4;
                ((float4*)op)[1] = hi4;
            }
        }
    }

    // --- stats: reduce across the 16 lanes that share a channel set ---
#pragma unroll
    for (int mt = 0; mt < MT; ++mt) {
#pragma unroll
        for (int r = 0; r < 8; ++r) {
            float s = s_sum[mt][r], q = s_sq[mt][r];
#pragma unroll
            for (int mask = 1; mask < 16; mask <<= 1) {
                s += __shfl_xor(s, mask);
                q += __shfl_xor(q, mask);
            }
            const int ch = mt * 16 + r + (hi ? 8 : 0);
            if (l15 == 0 && ch < MOUT) {
                atomicAdd(&sums[ch], s);
                atomicAdd(&sqsums[ch], q);
            }
        }
    }
}

// ---------------------------------------------------------------------------
// Output: zero diagonal, then symmetric scatter of BN4(h3)
// ---------------------------------------------------------------------------
__global__ void diag_zero_kernel(float* __restrict__ y) {
    int idx = blockIdx.x * 256 + threadIdx.x;            // B*8*S = 65536
    if (idx < Bdim * 8 * Sdim) {
        int s = idx & (Sdim - 1);
        int bc = idx >> 7;
        y[((size_t)bc * Sdim + s) * Sdim + s] = 0.0f;
    }
}

__global__ __launch_bounds__(256)
void scatter_kernel(const float* __restrict__ h3,       // [N][8]
                    const float* __restrict__ scale,    // [8]
                    const float* __restrict__ shift,    // [8]
                    float* __restrict__ y) {            // [B][8][S][S]
    const int idx = blockIdx.x * 256 + threadIdx.x;      // 0 .. NCOLS-1
    const int b = idx / NPAIR;
    const int p = idx - b * NPAIR;
    int i, j; pair_ij(p, i, j);

    const float4* hp = (const float4*)(h3 + (size_t)idx * 8);
    float4 a4 = hp[0], b4 = hp[1];
    float hv[8] = {a4.x, a4.y, a4.z, a4.w, b4.x, b4.y, b4.z, b4.w};
#pragma unroll
    for (int c = 0; c < 8; ++c) {
        const float v = hv[c] * scale[c] + shift[c];
        float* yb = y + ((size_t)b * 8 + c) * Sdim * Sdim;
        yb[(size_t)i * Sdim + j] = v;
        yb[(size_t)j * Sdim + i] = v;
    }
}

// ---------------------------------------------------------------------------
extern "C" void kernel_launch(void* const* d_in, const int* in_sizes, int n_in,
                              void* d_out, int out_size, void* d_ws, size_t ws_size,
                              hipStream_t stream) {
    auto F = [&](int k) { return (const float*)d_in[k]; };

    const float *X, *bn0g, *bn0b;
    const float *W[4], *Bv[4], *G[4], *Bt[4];
    if (n_in >= 19 && in_sizes[0] == Bdim * 4 * Sdim) {
        // dict insertion order: x, bn0_g, bn0_b, then blocks (W, b, g, bt)
        X = F(0); bn0g = F(1); bn0b = F(2);
        int k = 3;
        for (int n = 0; n < 4; ++n) { W[n] = F(k++); Bv[n] = F(k++); G[n] = F(k++); Bt[n] = F(k++); }
    } else {
        // sorted pytree flatten: blocks (W, b, bt, g) x4, bn0_b, bn0_g, x
        int k = 0;
        for (int n = 0; n < 4; ++n) { W[n] = F(k++); Bv[n] = F(k++); Bt[n] = F(k++); G[n] = F(k++); }
        bn0b = F(16); bn0g = F(17); X = F(18);
    }

    // workspace layout (floats):
    //   [5 BN slots x 256]: +0 sum[64], +64 sumsq[64], +128 scale[64], +192 shift[64]
    //   feats  [N][4]
    //   bufA   [N][64]   (h0, reused for h2)
    //   bufB   [N][64]   (h1, reused for h3 [N][8])
    float* wsf = (float*)d_ws;
    auto SUM = [&](int k) { return wsf + k * 256 + 0;   };
    auto SQ  = [&](int k) { return wsf + k * 256 + 64;  };
    auto SC  = [&](int k) { return wsf + k * 256 + 128; };
    auto SH  = [&](int k) { return wsf + k * 256 + 192; };
    float* feats = wsf + 2048;
    float* bufA  = feats + (size_t)4 * NCOLS;
    float* bufB  = bufA + (size_t)64 * NCOLS;

    const float invN = 1.0f / (float)NCOLS;
    const int NBLK = NCOLS / 256;   // 2032, exact
    const int GEMM_BLOCKS = 1024;

    zero_ws_kernel<<<8, 256, 0, stream>>>(wsf, 2048);

    feats_kernel<<<NBLK, 256, 0, stream>>>(X, feats, SUM(0), SQ(0));
    finalize_bn_kernel<<<1, 64, 0, stream>>>(SUM(0), SQ(0), bn0g, bn0b, SC(0), SH(0), 4, invN);

    gemm_bn_stats_kernel<4, false, 64><<<GEMM_BLOCKS, 256, 0, stream>>>(
        W[0], Bv[0], SC(0), SH(0), feats, bufA, SUM(1), SQ(1));
    finalize_bn_kernel<<<1, 64, 0, stream>>>(SUM(1), SQ(1), G[0], Bt[0], SC(1), SH(1), 64, invN);

    gemm_bn_stats_kernel<64, true, 64><<<GEMM_BLOCKS, 256, 0, stream>>>(
        W[1], Bv[1], SC(1), SH(1), bufA, bufB, SUM(2), SQ(2));
    finalize_bn_kernel<<<1, 64, 0, stream>>>(SUM(2), SQ(2), G[1], Bt[1], SC(2), SH(2), 64, invN);

    gemm_bn_stats_kernel<64, true, 64><<<GEMM_BLOCKS, 256, 0, stream>>>(
        W[2], Bv[2], SC(2), SH(2), bufB, bufA, SUM(3), SQ(3));
    finalize_bn_kernel<<<1, 64, 0, stream>>>(SUM(3), SQ(3), G[2], Bt[2], SC(3), SH(3), 64, invN);

    gemm_bn_stats_kernel<64, true, 8><<<GEMM_BLOCKS, 256, 0, stream>>>(
        W[3], Bv[3], SC(3), SH(3), bufA, bufB, SUM(4), SQ(4));
    finalize_bn_kernel<<<1, 64, 0, stream>>>(SUM(4), SQ(4), G[3], Bt[3], SC(4), SH(4), 8, invN);

    diag_zero_kernel<<<256, 256, 0, stream>>>((float*)d_out);
    scatter_kernel<<<NBLK, 256, 0, stream>>>(bufB, SC(4), SH(4), (float*)d_out);
}
